// ISTFTLayer_79791902425567
// MI455X (gfx1250) — matrix operand order repaired and checked by reading, not compiled, not run on hardware
//
#include <hip/hip_runtime.h>
#include <hip/hip_bf16.h>
#include <stddef.h>

// ---------------------------------------------------------------------------
// ISTFT as windowed-DFT GEMM + fused overlap-add on CDNA5 WMMA.
// bf16 hi/lo split (3 WMMAs per K-chunk: hi*hi + hi*lo + lo*hi), f32 accum.
// A is staged once per block into LDS (pre-split, pre-guarded, combined
// real|imag K axis of 1088); the hot loop is ds_load_b128 + global_load_b128
// + v_wmma only.  Overlap-add (WIN/HOP = 4) is fused as 4 row/col-shifted
// GEMM passes, so every output element is owned by exactly one wave.
// ---------------------------------------------------------------------------

#define WIN      1024
#define HOP      256
#define FRAMES   1024
#define BATCH    16
#define BINS     513
#define KPART    544                    // 513 padded to 17 * 32 (per part)
#define KTOT     1088                   // real 0..543 | imag 544..1087
#define KCHUNKS  34                     // KTOT / 32
#define SEGMENTS 1027                   // OUTPUT_LENGTH / HOP
#define OUTLEN   262912                 // SEGMENTS * HOP
#define SEGTILES 65                     // ceil(1027 / 16)
#define AROWS    19                     // frames s0-3 .. s0+15

typedef __attribute__((ext_vector_type(16))) __bf16          v16bf;
typedef __attribute__((ext_vector_type(8)))  float           v8f;
typedef __attribute__((ext_vector_type(8)))  unsigned short  u16x8;

union BF16x16 { v16bf v; u16x8 half2[2]; };

static __device__ __forceinline__ unsigned short f32_to_bf16_rn(float x) {
    unsigned u = __float_as_uint(x);
    unsigned r = u + 0x7FFFu + ((u >> 16) & 1u);   // round-to-nearest-even
    return (unsigned short)(r >> 16);
}
static __device__ __forceinline__ float bf16_bits_to_f32(unsigned short h) {
    return __uint_as_float(((unsigned)h) << 16);
}

// ---------------------------------------------------------------------------
// Kernel 1: combined basis plane in workspace, layout [n=0..1023][k=0..1087],
// k<544 -> windowed cos basis (real part), k>=544 -> windowed -sin basis
// (imag part); k pads zeroed.  hi/lo bf16 planes: 2 * 1024*1088*2B = 4.45 MB.
// ---------------------------------------------------------------------------
__global__ __launch_bounds__(256) void istft_basis_kernel(
    unsigned short* __restrict__ bas_hi, unsigned short* __restrict__ bas_lo) {
    int t = blockIdx.x * blockDim.x + threadIdx.x;
    if (t >= WIN * KTOT) return;
    int n = t / KTOT;
    int k = t % KTOT;
    bool isReal = (k < KPART);
    int kk = isReal ? k : k - KPART;

    float v = 0.0f;
    if (kk < BINS) {
        // exact argument reduction: angle = 2*pi*((k*n) mod 1024)/1024
        int m = (kk * n) & (WIN - 1);
        float ang = (float)m * (6.28318530717958647692f / (float)WIN);
        float s, c;
        sincosf(ang, &s, &c);
        float wnd = 0.5f - 0.5f * cosf((float)n * (6.28318530717958647692f / (float)WIN));
        float scale = wnd * (1.0f / (float)WIN);
        bool edge = (kk == 0) || (kk == WIN / 2);
        if (isReal) v = scale * (edge ? 1.0f : 2.0f) * c;
        else        v = edge ? 0.0f : -scale * 2.0f * s;  // irfft drops imag @ DC/Nyq
    }
    unsigned short hb = f32_to_bf16_rn(v);
    bas_hi[t] = hb;
    bas_lo[t] = f32_to_bf16_rn(v - bf16_bits_to_f32(hb));
}

// ---------------------------------------------------------------------------
// Kernel 2: fused GEMM + overlap-add.
// Block = 256 threads = 8 waves -> (batch b, 16-segment tile s0).
// Stage: 19 frame rows x 1088 K, split to bf16 hi/lo in LDS (82.7 KB).
// Wave w owns output columns [w*32, w*32+32) as two 16x16 WMMA tiles.
// j = 0..3: rows are frames f = s0 + r - j (LDS row r+3-j, OOB rows zeroed),
//           basis columns n = j*256 + c.
// ---------------------------------------------------------------------------
__global__ __launch_bounds__(256) void istft_wmma_kernel(
    const float* __restrict__ xr, const float* __restrict__ xi,
    const unsigned short* __restrict__ bas_hi,
    const unsigned short* __restrict__ bas_lo,
    float* __restrict__ out) {

    __shared__ __attribute__((aligned(16))) unsigned short sAhi[AROWS * KTOT];
    __shared__ __attribute__((aligned(16))) unsigned short sAlo[AROWS * KTOT];

    const int tid  = threadIdx.x;
    const int lane = tid & 31;
    const int wave = tid >> 5;
    const int r    = lane & 15;        // A: row M ; B/C: col N
    const int h    = lane >> 4;        // half selector (K / M layout)
    const int s0   = blockIdx.x * 16;  // first output segment of this tile
    const int b    = blockIdx.y;
    const int c0   = wave * 32;        // two 16-wide column tiles: c0, c0+16

    // ---- cooperative A stage: load, guard, split once per block ----
    for (int idx = tid; idx < AROWS * KTOT; idx += 256) {
        int row = idx / KTOT;          // 0..18  -> frame f = s0 - 3 + row
        int k   = idx - row * KTOT;    // 0..1087
        int f   = s0 - 3 + row;
        bool isReal = (k < KPART);
        int  kk     = isReal ? k : k - KPART;
        const float* src = isReal ? xr : xi;
        bool ok = (f >= 0) && (f < FRAMES) && (kk < BINS);
        int  fr = (f >= 0 && f < FRAMES) ? f : 0;
        float x = ok ? src[((size_t)b * FRAMES + fr) * BINS + kk] : 0.0f;
        unsigned short hb = f32_to_bf16_rn(x);
        sAhi[idx] = hb;
        sAlo[idx] = f32_to_bf16_rn(x - bf16_bits_to_f32(hb));
    }
    __syncthreads();

    v8f acc0 = {};
    v8f acc1 = {};

    for (int j = 0; j < 4; ++j) {
        const int lrow = r + 3 - j;                    // LDS row for this lane
        const unsigned short* ah = sAhi + lrow * KTOT;
        const unsigned short* al = sAlo + lrow * KTOT;
        const int n0 = j * HOP + c0;                   // basis column base

        for (int kc = 0; kc < KCHUNKS; ++kc) {
            const int k0 = kc * 32;

            // ---- A fragment from LDS: two 16B chunks per plane ----
            // elems 0..7  -> K = k0 + 16h + e       (chunk at k0 + 16h)
            // elems 8..15 -> K = k0 + 16h + e       (chunk at k0 + 16h + 8)
            BF16x16 ahi, alo;
            ahi.half2[0] = *reinterpret_cast<const u16x8*>(ah + k0 + 16 * h);
            ahi.half2[1] = *reinterpret_cast<const u16x8*>(ah + k0 + 16 * h + 8);
            alo.half2[0] = *reinterpret_cast<const u16x8*>(al + k0 + 16 * h);
            alo.half2[1] = *reinterpret_cast<const u16x8*>(al + k0 + 16 * h + 8);

            // ---- B fragments from L2-resident basis ----
            {
                size_t bidx = (size_t)(n0 + r) * KTOT + (size_t)k0 + h * 16;
                v16bf Bh = *reinterpret_cast<const v16bf*>(bas_hi + bidx);
                v16bf Bl = *reinterpret_cast<const v16bf*>(bas_lo + bidx);
                acc0 = __builtin_amdgcn_wmma_f32_16x16x32_bf16(false, ahi.v, false, Bh,
                                                               (short)0, acc0, false, false);
                acc0 = __builtin_amdgcn_wmma_f32_16x16x32_bf16(false, ahi.v, false, Bl,
                                                               (short)0, acc0, false, false);
                acc0 = __builtin_amdgcn_wmma_f32_16x16x32_bf16(false, alo.v, false, Bh,
                                                               (short)0, acc0, false, false);
            }
            {
                size_t bidx = (size_t)(n0 + 16 + r) * KTOT + (size_t)k0 + h * 16;
                v16bf Bh = *reinterpret_cast<const v16bf*>(bas_hi + bidx);
                v16bf Bl = *reinterpret_cast<const v16bf*>(bas_lo + bidx);
                acc1 = __builtin_amdgcn_wmma_f32_16x16x32_bf16(false, ahi.v, false, Bh,
                                                               (short)0, acc1, false, false);
                acc1 = __builtin_amdgcn_wmma_f32_16x16x32_bf16(false, ahi.v, false, Bl,
                                                               (short)0, acc1, false, false);
                acc1 = __builtin_amdgcn_wmma_f32_16x16x32_bf16(false, alo.v, false, Bh,
                                                               (short)0, acc1, false, false);
            }
        }
    }

    // ---- store C: VGPR i -> segment s0 + i + 8h, column c0 + r ----
#pragma unroll
    for (int i = 0; i < 8; ++i) {
        int s = s0 + i + h * 8;
        if (s < SEGMENTS) {
            size_t o = (size_t)b * OUTLEN + (size_t)s * HOP;
            out[o + c0 + r]      = acc0[i];
            out[o + c0 + 16 + r] = acc1[i];
        }
    }
}

// ---------------------------------------------------------------------------
extern "C" void kernel_launch(void* const* d_in, const int* in_sizes, int n_in,
                              void* d_out, int out_size, void* d_ws, size_t ws_size,
                              hipStream_t stream) {
    (void)in_sizes; (void)n_in; (void)out_size; (void)ws_size;
    const float* xr = (const float*)d_in[0];
    const float* xi = (const float*)d_in[1];
    float* out = (float*)d_out;

    unsigned short* ws = (unsigned short*)d_ws;
    const size_t plane = (size_t)WIN * KTOT;           // 1,114,112 elems / plane
    unsigned short* bas_hi = ws;
    unsigned short* bas_lo = ws + plane;

    const int btotal = WIN * KTOT;
    istft_basis_kernel<<<(btotal + 255) / 256, 256, 0, stream>>>(bas_hi, bas_lo);

    dim3 grid(SEGTILES, BATCH);                        // 65 x 16 blocks, 8 waves
    istft_wmma_kernel<<<grid, 256, 0, stream>>>(xr, xi, bas_hi, bas_lo, out);
}
